// Attention_org_45758581571643
// MI455X (gfx1250) — matrix operand order repaired and checked by reading, not compiled, not run on hardware
//
#include <hip/hip_runtime.h>
#include <hip/hip_bf16.h>
#include <math.h>

// ---------------- problem constants ----------------
#define BDIM   4
#define TDIM   8
#define CDIM   256
#define NPATCH 196
#define HDIM   4
#define SDIM   (TDIM * NPATCH)      // 1568 = 49*32
#define BH     (BDIM * HDIM)        // 16
#define SCALE_QK 0.02525381361380526f   // 1/sqrt(1568)
#define EPS_IN   1e-5f

// ---------------- WMMA types ----------------
typedef __attribute__((ext_vector_type(16))) _Float16 v16h;
typedef __attribute__((ext_vector_type(8)))  float    v8f;
typedef __attribute__((ext_vector_type(2)))  _Float16 h2;

union U32H2 { unsigned int u; h2 h; };

// ---------------- GEMM tiling ----------------
// 256 threads = 8 waves (wave32). Block tile 128x64, wave tile 32x32 (2x2 WMMA).
// Double-buffered LDS staging; A tiles via async DMA (global_load_async_to_lds_b128).
#define TM 128
#define TN 64
#define TK 32
// Padded LDS row pitch in halves: 40 halves = 80B rows -> 16B-aligned B128 async
// stores, and row*20 DWORDs mod 64 banks is conflict-free across 16 rows.
#define APITCH 40
#define ABUF (TM * APITCH)          // 5120 halves
#define BBUF (TN * APITCH)          // 2560 halves
#define SHHALVES (2 * (ABUF + BBUF))

// low 32 bits of a generic pointer to __shared__ = LDS address
__device__ __forceinline__ unsigned lds_addr32(const void* p)
{
  return (unsigned)(unsigned long long)p;
}

// ---- A tile loaders (TM x TK halves, row-major into As[row*APITCH + k]) ----

// Full tile: CDNA5 async DMA global->LDS, 16B per lane, 2 issues per wave.
__device__ __forceinline__ void load_a_async(
    const _Float16* __restrict__ A, int lda, int m0, int k0,
    _Float16* As, int tid)
{
  const int wave = tid >> 5, lane = tid & 31;
#pragma unroll
  for (int q = 0; q < 2; ++q) {
    int c     = wave * 2 + q;          // chunk: 8 rows of 64B
    int row   = c * 8 + (lane >> 2);
    int piece = lane & 3;              // 16B pieces within the 64B row
    const _Float16* gp = A + (size_t)(m0 + row) * lda + k0 + piece * 8;
    unsigned ldst = lds_addr32(As + row * APITCH + piece * 8);
    asm volatile("global_load_async_to_lds_b128 %0, %1, off"
                 :: "v"(ldst), "v"((unsigned long long)(size_t)gp)
                 : "memory");
  }
}

// Edge tile: branch-free guarded loads (clamp address, select zero).
__device__ __forceinline__ void load_a_guard(
    const _Float16* __restrict__ A, int lda, int M, int m0, int k0,
    _Float16* As, int tid)
{
#pragma unroll
  for (int u = tid; u < (TM * TK / 2); u += 256) {
    int row = u >> 4;
    int kp  = (u & 15) << 1;
    int gm  = m0 + row;
    int cm  = (gm < M) ? gm : (M - 1);
    unsigned val = *(const unsigned int*)(A + (size_t)cm * lda + k0 + kp);
    val = (gm < M) ? val : 0u;
    *(unsigned int*)(As + row * APITCH + kp) = val;
  }
}

// ---- B tile: fetch (global->regs) / commit (regs->LDS transposed) ----

struct BReg { unsigned v[4]; };

__device__ __forceinline__ BReg b_fetch(
    const _Float16* __restrict__ Bm, int ldb, int N, int n0, int k0,
    bool bfull, int tid)
{
  BReg r;
  if (bfull) {
    int k  = tid >> 3;           // 0..31
    int np = (tid & 7) * 8;      // 0..56
    uint4 x = *(const uint4*)(Bm + (size_t)(k0 + k) * ldb + n0 + np);
    r.v[0] = x.x; r.v[1] = x.y; r.v[2] = x.z; r.v[3] = x.w;
  } else {
#pragma unroll
    for (int j = 0; j < 4; ++j) {
      int u  = tid + j * 256;
      int k  = u >> 5;
      int np = (u & 31) << 1;
      int gn = n0 + np;
      int cn = (gn < N) ? gn : (N - 2);
      unsigned val = *(const unsigned int*)(Bm + (size_t)(k0 + k) * ldb + cn);
      r.v[j] = (gn < N) ? val : 0u;
    }
  }
  return r;
}

__device__ __forceinline__ void b_commit(const BReg& r, _Float16* BsT, bool bfull, int tid)
{
  U32H2 t;
  if (bfull) {
    int k  = tid >> 3;
    int np = (tid & 7) * 8;
    t.u = r.v[0]; BsT[(np + 0) * APITCH + k] = t.h[0]; BsT[(np + 1) * APITCH + k] = t.h[1];
    t.u = r.v[1]; BsT[(np + 2) * APITCH + k] = t.h[0]; BsT[(np + 3) * APITCH + k] = t.h[1];
    t.u = r.v[2]; BsT[(np + 4) * APITCH + k] = t.h[0]; BsT[(np + 5) * APITCH + k] = t.h[1];
    t.u = r.v[3]; BsT[(np + 6) * APITCH + k] = t.h[0]; BsT[(np + 7) * APITCH + k] = t.h[1];
  } else {
#pragma unroll
    for (int j = 0; j < 4; ++j) {
      int u  = tid + j * 256;
      int k  = u >> 5;
      int np = (u & 31) << 1;
      t.u = r.v[j];
      BsT[(np    ) * APITCH + k] = t.h[0];
      BsT[(np + 1) * APITCH + k] = t.h[1];
    }
  }
}

// A fragment 16x32 f16 (ISA 7.12.2): lanes 0-15 row M=lane, K 0..7 in v0..3,
// K 16..23 in v4..7; lanes 16-31 same rows, K 8..15 / 24..31.
__device__ __forceinline__ v16h load_afrag(const _Float16* As, int rowbase, int lane)
{
  const int hh = (lane >> 4) & 1;
  const int m  = lane & 15;
  const _Float16* rp = As + (rowbase + m) * APITCH;
  v16h a;
#pragma unroll
  for (int v = 0; v < 8; ++v) {
    int k = (v < 4) ? (2 * v + 8 * hh) : (2 * v + 8 + 8 * hh);  // v>=4: 16+2(v-4)
    h2 p = *(const h2*)(rp + k);
    a[2 * v]     = p[0];
    a[2 * v + 1] = p[1];
  }
  return a;
}

// B fragment 32x16 f16: lane = column n; lanes 0-15 hold K 0..7 / 16..23 (2/VGPR),
// lanes 16-31 hold K 8..15 / 24..31 (same k order as the A fragment).
__device__ __forceinline__ v16h load_bfrag(const _Float16* BsT, int colbase, int lane)
{
  const int hh = (lane >> 4) & 1;
  const int n  = lane & 15;
  const _Float16* cp = BsT + (colbase + n) * APITCH;
  v16h b;
#pragma unroll
  for (int v = 0; v < 8; ++v) {
    int k = (v < 4) ? (2 * v + 8 * hh) : (2 * v + 8 + 8 * hh);
    h2 p = *(const h2*)(cp + k);
    b[2 * v]     = p[0];
    b[2 * v + 1] = p[1];
  }
  return b;
}

// Core k-loop, double-buffered; accumulates into acc (callers may run it
// repeatedly, e.g. over heads). K compile-time -> static unroll for K=256.
template <int K>
__device__ __forceinline__ void gemm_acc(
    const _Float16* __restrict__ A, int lda, int M, int m0,
    const _Float16* __restrict__ Bm, int ldb, int N, int n0,
    _Float16* SH, v8f acc[2][2])
{
  constexpr int NK = K / TK;
  const int tid  = threadIdx.x;
  const int lane = tid & 31;
  const int wave = tid >> 5;
  const int wm = (wave >> 1) * 32;
  const int wn = (wave & 1) * 32;
  const bool afull = (m0 + TM <= M);   // block-uniform
  const bool bfull = (n0 + TN <= N);   // block-uniform

  __syncthreads();   // protect prologue overwrite across repeated calls
  // prologue: stage tile 0 into buffer 0
  if (afull) load_a_async(A, lda, m0, 0, SH, tid);
  else       load_a_guard(A, lda, M, m0, 0, SH, tid);
  {
    BReg br = b_fetch(Bm, ldb, N, n0, 0, bfull, tid);
    b_commit(br, SH + ABUF, bfull, tid);
  }

#pragma unroll
  for (int s = 0; s < NK; ++s) {
    const int p = s & 1;
    _Float16* Ac = SH + p * (ABUF + BBUF);
    _Float16* Bc = Ac + ABUF;
    _Float16* An = SH + (p ^ 1) * (ABUF + BBUF);
    _Float16* Bn = An + ABUF;

    if (afull)
      asm volatile("s_wait_asynccnt 0" ::: "memory");
    __syncthreads();   // tile s fully in LDS; prior reads of buffers done

    const bool have_next = (s + 1 < NK);
    BReg br;
    if (have_next) {
      const int k0 = (s + 1) * TK;
      if (afull) load_a_async(A, lda, m0, k0, An, tid);   // DMA overlaps WMMA
      else       load_a_guard(A, lda, M, m0, k0, An, tid);
      br = b_fetch(Bm, ldb, N, n0, k0, bfull, tid);       // in flight during WMMA
    }

    v16h a0 = load_afrag(Ac, wm,      lane);
    v16h a1 = load_afrag(Ac, wm + 16, lane);
    v16h b0 = load_bfrag(Bc, wn,      lane);
    v16h b1 = load_bfrag(Bc, wn + 16, lane);
    acc[0][0] = __builtin_amdgcn_wmma_f32_16x16x32_f16(false, a0, false, b0, (short)0, acc[0][0], false, false);
    acc[0][1] = __builtin_amdgcn_wmma_f32_16x16x32_f16(false, a0, false, b1, (short)0, acc[0][1], false, false);
    acc[1][0] = __builtin_amdgcn_wmma_f32_16x16x32_f16(false, a1, false, b0, (short)0, acc[1][0], false, false);
    acc[1][1] = __builtin_amdgcn_wmma_f32_16x16x32_f16(false, a1, false, b1, (short)0, acc[1][1], false, false);

    if (have_next)
      b_commit(br, Bn, bfull, tid);    // LDS stores after WMMA issue
  }
}

// ---------------- kernels ----------------

// emb [B][T][C][N] f32 -> xh [B][T*N][C] f16
__global__ void k_prep_x(const float* __restrict__ emb, _Float16* __restrict__ xh)
{
  int id = blockIdx.x * 256 + threadIdx.x;
  if (id >= BDIM * TDIM * CDIM * NPATCH) return;
  int n = id % NPATCH;
  int c = (id / NPATCH) % CDIM;
  int t = (id / (NPATCH * CDIM)) % TDIM;
  int b = id / (NPATCH * CDIM * TDIM);
  xh[((size_t)b * SDIM + t * NPATCH + n) * CDIM + c] = (_Float16)emb[id];
}

// Wq/Wk/Wv [H][D][C] -> [H][C][D] f16 (B-operand K x N layout); Wo likewise.
__global__ void k_prep_w(const float* __restrict__ wq, const float* __restrict__ wk,
                         const float* __restrict__ wv, const float* __restrict__ wo,
                         _Float16* __restrict__ wqt, _Float16* __restrict__ wkt,
                         _Float16* __restrict__ wvt, _Float16* __restrict__ wot)
{
  int z  = blockIdx.z;
  int id = blockIdx.x * 256 + threadIdx.x;
  if (z < 3) {
    if (id >= HDIM * CDIM * CDIM) return;
    int c = id % CDIM;
    int d = (id / CDIM) % CDIM;
    int h = id / (CDIM * CDIM);
    const float* src = (z == 0) ? wq : (z == 1) ? wk : wv;
    _Float16*    dst = (z == 0) ? wqt : (z == 1) ? wkt : wvt;
    dst[((size_t)h * CDIM + c) * CDIM + d] = (_Float16)src[id];
  } else {
    if (id >= CDIM * CDIM) return;
    int c = id % CDIM;
    int d = id / CDIM;
    wot[(size_t)c * CDIM + d] = (_Float16)wo[id];
  }
}

__global__ void k_init_stats(float* g)
{
  int i = threadIdx.x;
  if (i < 64) g[i] = 0.f;
}

// QKV projections. blockIdx.z = which*16 + (b*H+h). K stored transposed [C][S].
__global__ __launch_bounds__(256) void k_proj(
    const _Float16* __restrict__ xh,
    const _Float16* __restrict__ wqt, const _Float16* __restrict__ wkt,
    const _Float16* __restrict__ wvt,
    _Float16* __restrict__ qh, _Float16* __restrict__ kht, _Float16* __restrict__ vh)
{
  __shared__ _Float16 SH[SHHALVES];
  const int m0 = blockIdx.x * TM;
  const int n0 = blockIdx.y * TN;
  const int which = blockIdx.z / BH;
  const int bh    = blockIdx.z % BH;
  const int b = bh / HDIM, h = bh % HDIM;

  const _Float16* A = xh + (size_t)b * SDIM * CDIM;
  const _Float16* W = ((which == 0) ? wqt : (which == 1) ? wkt : wvt) + (size_t)h * CDIM * CDIM;

  v8f acc[2][2];
  v8f zero8 = {};
#pragma unroll
  for (int i = 0; i < 2; ++i)
#pragma unroll
    for (int j = 0; j < 2; ++j) acc[i][j] = zero8;

  gemm_acc<CDIM>(A, CDIM, SDIM, m0, W, CDIM, CDIM, n0, SH, acc);

  const int lane = threadIdx.x & 31, wave = threadIdx.x >> 5;
  const int wm = (wave >> 1) * 32, wn = (wave & 1) * 32;
  const int rl = (lane >> 4) * 8, cl = lane & 15;
#pragma unroll
  for (int i = 0; i < 2; ++i)
#pragma unroll
    for (int j = 0; j < 2; ++j)
#pragma unroll
      for (int r = 0; r < 8; ++r) {
        int m = m0 + wm + i * 16 + r + rl;
        int n = n0 + wn + j * 16 + cl;
        if (m < SDIM) {
          _Float16 v = (_Float16)acc[i][j][r];
          if (which == 1)      kht[((size_t)bh * CDIM + n) * SDIM + m] = v;
          else if (which == 0) qh [((size_t)bh * SDIM + m) * CDIM + n] = v;
          else                 vh [((size_t)bh * SDIM + m) * CDIM + n] = v;
        }
      }
}

// scores = Q K^T / sqrt(S); store f16; block-reduced sum/sum^2 -> atomics.
__global__ __launch_bounds__(256) void k_scores(
    const _Float16* __restrict__ qh, const _Float16* __restrict__ kht,
    _Float16* __restrict__ sc, float* __restrict__ gsum, float* __restrict__ gsumsq)
{
  __shared__ _Float16 SH[SHHALVES];
  __shared__ float reds[256];
  __shared__ float redq[256];
  const int m0 = blockIdx.x * TM;
  const int n0 = blockIdx.y * TN;
  const int bh = blockIdx.z;

  v8f acc[2][2];
  v8f zero8 = {};
#pragma unroll
  for (int i = 0; i < 2; ++i)
#pragma unroll
    for (int j = 0; j < 2; ++j) acc[i][j] = zero8;

  gemm_acc<CDIM>(qh + (size_t)bh * SDIM * CDIM, CDIM, SDIM, m0,
                 kht + (size_t)bh * CDIM * SDIM, SDIM, SDIM, n0,
                 SH, acc);

  const int tid = threadIdx.x;
  const int lane = tid & 31, wave = tid >> 5;
  const int wm = (wave >> 1) * 32, wn = (wave & 1) * 32;
  const int rl = (lane >> 4) * 8, cl = lane & 15;
  float ls = 0.f, lq = 0.f;
#pragma unroll
  for (int i = 0; i < 2; ++i)
#pragma unroll
    for (int j = 0; j < 2; ++j)
#pragma unroll
      for (int r = 0; r < 8; ++r) {
        int m = m0 + wm + i * 16 + r + rl;
        int n = n0 + wn + j * 16 + cl;
        if (m < SDIM && n < SDIM) {
          float v = acc[i][j][r] * SCALE_QK;
          sc[((size_t)bh * SDIM + m) * SDIM + n] = (_Float16)v;
          ls += v;
          lq += v * v;
        }
      }
  reds[tid] = ls; redq[tid] = lq;
  __syncthreads();
  for (int s = 128; s > 0; s >>= 1) {
    if (tid < s) { reds[tid] += reds[tid + s]; redq[tid] += redq[tid + s]; }
    __syncthreads();
  }
  if (tid == 0) {
    atomicAdd(&gsum[bh], reds[0]);
    atomicAdd(&gsumsq[bh], redq[0]);
  }
}

__global__ void k_finalize(const float* gsum, const float* gsumsq, float* mean, float* inv)
{
  int i = threadIdx.x;
  if (i < BH) {
    float cnt = (float)SDIM * (float)SDIM;
    float m = gsum[i] / cnt;
    float v = gsumsq[i] / cnt - m * m;
    mean[i] = m;
    inv[i]  = rsqrtf(v + EPS_IN);
  }
}

// Per row: instance-norm, softmax; store probs/(H*rowsum) in place (f16).
__global__ __launch_bounds__(256) void k_softmax(
    _Float16* __restrict__ sc, const float* __restrict__ mean, const float* __restrict__ inv)
{
  __shared__ float rowf[SDIM];
  __shared__ float red[256];
  const int s = blockIdx.x, bh = blockIdx.y, tid = threadIdx.x;
  _Float16* row = sc + ((size_t)bh * SDIM + s) * SDIM;
  const float mu = mean[bh], iv = inv[bh];

  float lmax = -3.0e38f;
  for (int i = tid; i < SDIM; i += 256) {
    float nv = ((float)row[i] - mu) * iv;
    rowf[i] = nv;
    lmax = fmaxf(lmax, nv);
  }
  red[tid] = lmax;
  __syncthreads();
  for (int st = 128; st > 0; st >>= 1) {
    if (tid < st) red[tid] = fmaxf(red[tid], red[tid + st]);
    __syncthreads();
  }
  float mx = red[0];
  __syncthreads();

  float lsum = 0.f;
  for (int i = tid; i < SDIM; i += 256) {
    float e = __expf(rowf[i] - mx);
    rowf[i] = e;
    lsum += e;
  }
  red[tid] = lsum;
  __syncthreads();
  for (int st = 128; st > 0; st >>= 1) {
    if (tid < st) red[tid] += red[tid + st];
    __syncthreads();
  }
  float scale = 1.0f / (red[0] * (float)HDIM);
  __syncthreads();
  for (int i = tid; i < SDIM; i += 256)
    row[i] = (_Float16)(rowf[i] * scale);
}

// ctx (head-mean folded): ctxm[b] = sum_h P[b,h] @ V[b,h]
__global__ __launch_bounds__(256) void k_ctx(
    const _Float16* __restrict__ sc, const _Float16* __restrict__ vh,
    _Float16* __restrict__ ctxm)
{
  __shared__ _Float16 SH[SHHALVES];
  const int m0 = blockIdx.x * TM;
  const int n0 = blockIdx.y * TN;
  const int b  = blockIdx.z;

  v8f acc[2][2];
  v8f zero8 = {};
#pragma unroll
  for (int i = 0; i < 2; ++i)
#pragma unroll
    for (int j = 0; j < 2; ++j) acc[i][j] = zero8;

  for (int h = 0; h < HDIM; ++h) {
    int bh = b * HDIM + h;
    gemm_acc<SDIM>(sc + (size_t)bh * SDIM * SDIM, SDIM, SDIM, m0,
                   vh + (size_t)bh * SDIM * CDIM, CDIM, CDIM, n0,
                   SH, acc);
  }

  const int lane = threadIdx.x & 31, wave = threadIdx.x >> 5;
  const int wm = (wave >> 1) * 32, wn = (wave & 1) * 32;
  const int rl = (lane >> 4) * 8, cl = lane & 15;
#pragma unroll
  for (int i = 0; i < 2; ++i)
#pragma unroll
    for (int j = 0; j < 2; ++j)
#pragma unroll
      for (int r = 0; r < 8; ++r) {
        int m = m0 + wm + i * 16 + r + rl;
        int n = n0 + wn + j * 16 + cl;
        if (m < SDIM)
          ctxm[((size_t)b * SDIM + m) * CDIM + n] = (_Float16)acc[i][j][r];
      }
}

// out[b] = ctxm[b] @ Wo^T, f32 stores straight into d_out (reshape is a view).
__global__ __launch_bounds__(256) void k_out(
    const _Float16* __restrict__ ctxm, const _Float16* __restrict__ wot,
    float* __restrict__ out)
{
  __shared__ _Float16 SH[SHHALVES];
  const int m0 = blockIdx.x * TM;
  const int n0 = blockIdx.y * TN;
  const int b  = blockIdx.z;

  v8f acc[2][2];
  v8f zero8 = {};
#pragma unroll
  for (int i = 0; i < 2; ++i)
#pragma unroll
    for (int j = 0; j < 2; ++j) acc[i][j] = zero8;

  gemm_acc<CDIM>(ctxm + (size_t)b * SDIM * CDIM, CDIM, SDIM, m0,
                 wot, CDIM, CDIM, n0, SH, acc);

  const int lane = threadIdx.x & 31, wave = threadIdx.x >> 5;
  const int wm = (wave >> 1) * 32, wn = (wave & 1) * 32;
  const int rl = (lane >> 4) * 8, cl = lane & 15;
#pragma unroll
  for (int i = 0; i < 2; ++i)
#pragma unroll
    for (int j = 0; j < 2; ++j)
#pragma unroll
      for (int r = 0; r < 8; ++r) {
        int m = m0 + wm + i * 16 + r + rl;
        int n = n0 + wn + j * 16 + cl;
        if (m < SDIM)
          out[((size_t)b * SDIM + m) * CDIM + n] = acc[i][j][r];
      }
}

// ---------------- host ----------------
static constexpr size_t align_up(size_t x) { return (x + 255) & ~(size_t)255; }

extern "C" void kernel_launch(void* const* d_in, const int* in_sizes, int n_in,
                              void* d_out, int out_size, void* d_ws, size_t ws_size,
                              hipStream_t stream)
{
  const float* emb = (const float*)d_in[0];
  const float* Wq  = (const float*)d_in[1];
  const float* Wk  = (const float*)d_in[2];
  const float* Wv  = (const float*)d_in[3];
  const float* Wo  = (const float*)d_in[4];
  float* out = (float*)d_out;

  // workspace layout (f16 element counts)
  constexpr size_t NXH  = (size_t)BDIM * SDIM * CDIM;          // x f16
  constexpr size_t NWT  = (size_t)HDIM * CDIM * CDIM;          // each W^T f16
  constexpr size_t NWO  = (size_t)CDIM * CDIM;
  constexpr size_t NP   = (size_t)BDIM * HDIM * SDIM * CDIM;   // Q / K^T / V
  constexpr size_t NSC  = (size_t)BDIM * HDIM * SDIM * SDIM;   // scores / probs
  constexpr size_t NCTX = (size_t)BDIM * SDIM * CDIM;

  size_t off = 0;
  char* ws = (char*)d_ws;
  _Float16* xh  = (_Float16*)(ws + off); off = align_up(off + NXH * 2);
  _Float16* wqt = (_Float16*)(ws + off); off = align_up(off + NWT * 2);
  _Float16* wkt = (_Float16*)(ws + off); off = align_up(off + NWT * 2);
  _Float16* wvt = (_Float16*)(ws + off); off = align_up(off + NWT * 2);
  _Float16* wot = (_Float16*)(ws + off); off = align_up(off + NWO * 2);
  _Float16* qh  = (_Float16*)(ws + off); off = align_up(off + NP * 2);
  _Float16* kht = (_Float16*)(ws + off); off = align_up(off + NP * 2);
  _Float16* vh  = (_Float16*)(ws + off); off = align_up(off + NP * 2);
  _Float16* sc  = (_Float16*)(ws + off); off = align_up(off + NSC * 2);
  _Float16* ctx = (_Float16*)(ws + off); off = align_up(off + NCTX * 2);
  float* stats  = (float*)(ws + off);    off = align_up(off + 64 * sizeof(float));
  float* gsum   = stats;
  float* gsumsq = stats + 16;
  float* meanv  = stats + 32;
  float* invv   = stats + 48;
  (void)ws_size; (void)in_sizes; (void)n_in; (void)out_size;

  const int GM = (SDIM + TM - 1) / TM;   // 13
  const int GNS = (SDIM + TN - 1) / TN;  // 25
  const int GNC = CDIM / TN;             // 4

  { int tot = BDIM * TDIM * CDIM * NPATCH;
    k_prep_x<<<(tot + 255) / 256, 256, 0, stream>>>(emb, xh); }
  { int tot = HDIM * CDIM * CDIM;
    k_prep_w<<<dim3((tot + 255) / 256, 1, 4), 256, 0, stream>>>(
        Wq, Wk, Wv, Wo, wqt, wkt, wvt, wot); }
  k_init_stats<<<1, 64, 0, stream>>>(stats);

  k_proj<<<dim3(GM, GNC, 3 * BH), 256, 0, stream>>>(xh, wqt, wkt, wvt, qh, kht, vh);
  k_scores<<<dim3(GM, GNS, BH), 256, 0, stream>>>(qh, kht, sc, gsum, gsumsq);
  k_finalize<<<1, 16, 0, stream>>>(gsum, gsumsq, meanv, invv);
  k_softmax<<<dim3(SDIM, BH), 256, 0, stream>>>(sc, meanv, invv);
  k_ctx<<<dim3(GM, GNC, BDIM), 256, 0, stream>>>(sc, vh, ctx);
  k_out<<<dim3(GM, GNC, BDIM), 256, 0, stream>>>(ctx, wot, out);
}
